// TopKGate_20160576487587
// MI455X (gfx1250) — compile-verified, matrix operand order checked by reading, not run on hardware
//
#include <hip/hip_runtime.h>
#include <stdint.h>

// ---------------------------------------------------------------------------
// MoE Top-K router gate for MI455X (gfx1250, wave32, WMMA).
//   logits = X @ W^T + noise/64 ; softmax ; top-8 ; aux load-balancing loss
// GEMM phase uses V_WMMA_F32_16X16X32_BF16 (f32 accumulate). X is streamed
// from HBM (268 MB -> kernel is HBM-bound, ~12us floor at 23.3 TB/s); W is
// pre-converted once to bf16 (512 KB, L2-resident).
// ---------------------------------------------------------------------------

typedef __attribute__((ext_vector_type(16))) __bf16 v16bf;
typedef __attribute__((ext_vector_type(8)))  float  v8f;

#define NTOK        16384
#define DIM         4096
#define NEXP        64
#define TOPK        8
#define ROWS_PER_WG 128     // 8 waves * 16-row WMMA tiles

union V16 {
    unsigned int u[8];
    v16bf        bf;
};

__device__ __forceinline__ unsigned int pack2_bf16(float lo, float hi) {
    // round-to-nearest-even f32 -> bf16, packed pair
    unsigned int a = __float_as_uint(lo);
    unsigned int b = __float_as_uint(hi);
    a = (a + 0x7FFFu + ((a >> 16) & 1u)) >> 16;
    b = (b + 0x7FFFu + ((b >> 16) & 1u)) >> 16;
    return (a & 0xFFFFu) | (b << 16);
}

// --- prologue: W f32 [64,4096] -> bf16 copy in workspace ---------------------
__global__ __launch_bounds__(512) void convert_w_kernel(const float* __restrict__ W,
                                                        unsigned int* __restrict__ Wbf_u32,
                                                        int npairs) {
    int i = blockIdx.x * blockDim.x + threadIdx.x;
    if (i < npairs) Wbf_u32[i] = pack2_bf16(W[2 * i], W[2 * i + 1]);
}

// --- main: GEMM (WMMA bf16) + softmax + top-8 + aux partials -----------------
__global__ __launch_bounds__(256) void router_kernel(
    const float*          __restrict__ X,      // [16384, 4096] f32
    const unsigned short* __restrict__ Wbf,    // [64, 4096] bf16
    const float*          __restrict__ noise,  // [16384, 64] f32
    float*                __restrict__ out,    // indices | weights | aux
    float*                __restrict__ red)    // [0..63]=counts [64..127]=score sums
{
    __shared__ float s_scores[ROWS_PER_WG * NEXP];  // 32 KB: logits -> scores
    __shared__ float s_cnt[NEXP];

    const int tid  = threadIdx.x;
    const int wave = tid >> 5;
    const int lane = tid & 31;
    const int half = lane >> 4;   // which 16-lane half (K-split in WMMA layout)
    const int m    = lane & 15;   // row of A / column of B within the tile

    if (tid < NEXP) s_cnt[tid] = 0.0f;

    // ---------------- Phase 1: logits tile via WMMA --------------------------
    const int rowBase = blockIdx.x * ROWS_PER_WG + wave * 16;
    const float* xrow = X + (size_t)(rowBase + m) * DIM;

    const unsigned short* wr0 = Wbf + (size_t)( 0 + m) * DIM;
    const unsigned short* wr1 = Wbf + (size_t)(16 + m) * DIM;
    const unsigned short* wr2 = Wbf + (size_t)(32 + m) * DIM;
    const unsigned short* wr3 = Wbf + (size_t)(48 + m) * DIM;

    v8f acc[4] = { v8f{}, v8f{}, v8f{}, v8f{} };

    for (int k = 0; k < DIM; k += 32) {
        const int kc0 = k + half * 8;        // K 0..7 (half 0) / 8..15 (half 1)
        const int kc1 = k + 16 + half * 8;   // K 16..23 / 24..31

        // A tile: 16x32 bf16, lane holds row m, 8+8 contiguous K elems
        float4 a0 = *(const float4*)(xrow + kc0);
        float4 a1 = *(const float4*)(xrow + kc0 + 4);
        float4 a2 = *(const float4*)(xrow + kc1);
        float4 a3 = *(const float4*)(xrow + kc1 + 4);
        V16 A;
        A.u[0] = pack2_bf16(a0.x, a0.y);  A.u[1] = pack2_bf16(a0.z, a0.w);
        A.u[2] = pack2_bf16(a1.x, a1.y);  A.u[3] = pack2_bf16(a1.z, a1.w);
        A.u[4] = pack2_bf16(a2.x, a2.y);  A.u[5] = pack2_bf16(a2.z, a2.w);
        A.u[6] = pack2_bf16(a3.x, a3.y);  A.u[7] = pack2_bf16(a3.z, a3.w);

        const unsigned short* wr[4] = { wr0, wr1, wr2, wr3 };
        #pragma unroll
        for (int t = 0; t < 4; ++t) {
            // B tile: 32x16 bf16; lane holds expert column m; W row-major means
            // a K-chunk of B is 8 contiguous bf16 of W's row -> single b128 load.
            uint4 q0 = *(const uint4*)(wr[t] + kc0);
            uint4 q1 = *(const uint4*)(wr[t] + kc1);
            V16 B;
            B.u[0] = q0.x; B.u[1] = q0.y; B.u[2] = q0.z; B.u[3] = q0.w;
            B.u[4] = q1.x; B.u[5] = q1.y; B.u[6] = q1.z; B.u[7] = q1.w;

            acc[t] = __builtin_amdgcn_wmma_f32_16x16x32_bf16(
                false, A.bf, false, B.bf, (short)0, acc[t], false, false);
        }
    }

    // C layout: VGPR r -> row (r + 8*half), column m, per 16-expert tile t
    #pragma unroll
    for (int t = 0; t < 4; ++t) {
        #pragma unroll
        for (int r = 0; r < 8; ++r) {
            s_scores[(wave * 16 + half * 8 + r) * NEXP + t * 16 + m] = acc[t][r];
        }
    }
    __syncthreads();

    // ---------------- Phase 2: per-row softmax (in LDS) ----------------------
    if (tid < ROWS_PER_WG) {
        const size_t gRow = (size_t)blockIdx.x * ROWS_PER_WG + tid;
        float* rowp = &s_scores[tid * NEXP];
        const float4* np = (const float4*)(noise + gRow * NEXP);

        float mx = -3.4e38f;
        #pragma unroll
        for (int e4 = 0; e4 < NEXP / 4; ++e4) {
            float4 nv = np[e4];
            float v0 = rowp[4 * e4 + 0] + nv.x * (1.0f / 64.0f);
            float v1 = rowp[4 * e4 + 1] + nv.y * (1.0f / 64.0f);
            float v2 = rowp[4 * e4 + 2] + nv.z * (1.0f / 64.0f);
            float v3 = rowp[4 * e4 + 3] + nv.w * (1.0f / 64.0f);
            rowp[4 * e4 + 0] = v0; rowp[4 * e4 + 1] = v1;
            rowp[4 * e4 + 2] = v2; rowp[4 * e4 + 3] = v3;
            mx = fmaxf(mx, fmaxf(fmaxf(v0, v1), fmaxf(v2, v3)));
        }
        float s = 0.0f;
        for (int e = 0; e < NEXP; ++e) {
            float v = __expf(rowp[e] - mx);
            rowp[e] = v;
            s += v;
        }
        const float inv = 1.0f / s;
        for (int e = 0; e < NEXP; ++e) rowp[e] *= inv;
    }
    __syncthreads();

    // ---------------- Phase 3: aux-loss score column sums --------------------
    if (tid < NEXP) {
        float cs = 0.0f;
        for (int r = 0; r < ROWS_PER_WG; ++r) cs += s_scores[r * NEXP + tid];
        atomicAdd(&red[NEXP + tid], cs);
    }
    __syncthreads();

    // ---------------- Phase 4: top-8 selection (mutates LDS scores) ----------
    if (tid < ROWS_PER_WG) {
        const size_t gRow = (size_t)blockIdx.x * ROWS_PER_WG + tid;
        float* rowp = &s_scores[tid * NEXP];
        for (int j = 0; j < TOPK; ++j) {
            float best = -1.0f;
            int   bi   = 0;
            for (int e = 0; e < NEXP; ++e) {
                float v = rowp[e];
                if (v > best) { best = v; bi = e; }   // ties -> lowest index
            }
            out[gRow * TOPK + j]                           = (float)bi;  // indices
            out[(size_t)NTOK * TOPK + gRow * TOPK + j]     = best;       // weights
            rowp[bi] = -1.0f;
            atomicAdd(&s_cnt[bi], 1.0f);
        }
    }
    __syncthreads();

    if (tid < NEXP) atomicAdd(&red[tid], s_cnt[tid]);
}

// --- epilogue: aux = 0.01 * E * sum_e (cnt_e/N) * (scoreSum_e/N) -------------
__global__ __launch_bounds__(64) void aux_kernel(const float* __restrict__ red,
                                                 float* __restrict__ aux_out) {
    __shared__ float r[NEXP];
    const int   e    = threadIdx.x;
    const float invN = 1.0f / (float)NTOK;
    r[e] = (red[e] * invN) * (red[NEXP + e] * invN);
    __syncthreads();
    if (e == 0) {
        float s = 0.0f;
        for (int i = 0; i < NEXP; ++i) s += r[i];
        aux_out[0] = 0.01f * (float)NEXP * s;
    }
}

extern "C" void kernel_launch(void* const* d_in, const int* in_sizes, int n_in,
                              void* d_out, int out_size, void* d_ws, size_t ws_size,
                              hipStream_t stream) {
    const float* X     = (const float*)d_in[0];   // [4,4096,4096] f32
    const float* W     = (const float*)d_in[1];   // [64,4096] f32
    const float* noise = (const float*)d_in[2];   // [16384,64] f32
    float* out = (float*)d_out;

    // workspace layout: [0,512) reduction accumulators (128 f32, zeroed each
    // call), [512, 512+512K) bf16 copy of W. Needs ~513 KB of d_ws.
    float*          red = (float*)d_ws;
    unsigned int*   Wbf = (unsigned int*)((char*)d_ws + 512);

    hipMemsetAsync(d_ws, 0, 512, stream);  // zero aux accumulators (capturable)

    convert_w_kernel<<<(NEXP * DIM / 2 + 511) / 512, 512, 0, stream>>>(
        W, Wbf, NEXP * DIM / 2);

    router_kernel<<<NTOK / ROWS_PER_WG, 256, 0, stream>>>(
        X, (const unsigned short*)Wbf, noise, out, red);

    aux_kernel<<<1, 64, 0, stream>>>(red, out + (size_t)NTOK * TOPK * 2);
}
